// GNN_Energy_Model_1st_Order_01_71347996721297
// MI455X (gfx1250) — matrix operand order repaired and checked by reading, not compiled, not run on hardware
//
#include <hip/hip_runtime.h>
#include <hip/hip_bf16.h>

// ---------------------------------------------------------------------------
// GNN-EBM forward for MI455X (gfx1250, wave32, WMMA bf16 w/ fp32 accumulate)
// B=8 T=512 M=4096 D=256, 3 layers, readout 512->512->256->1
// ---------------------------------------------------------------------------

#define BB 8
#define TT 512
#define MM 4096
#define DD 256

typedef __attribute__((ext_vector_type(16))) __bf16 v16bf;
typedef __attribute__((ext_vector_type(8)))  __bf16 bf16x8;
typedef __attribute__((ext_vector_type(8)))  float  v8f;

// ---- WMMA wrapper: D = A(16x32 bf16) * B(32x16 bf16) + C(16x16 f32) -------
__device__ __forceinline__ v8f wmma_bf16(v16bf a, v16bf b, v8f c) {
  return __builtin_amdgcn_wmma_f32_16x16x32_bf16(
      /*neg_a=*/false, a, /*neg_b=*/false, b,
      /*c_mod=*/(short)0, c, /*reuse_a=*/false, /*reuse_b=*/false);
}

// ---- A-fragment (16-bit A 16x32 layout): per-lane chunks at koff, koff+16 --
// koff = k0 + 8*(lane>>4). Source rows are fp32; optional SUM of two rows.
template <bool SUM>
__device__ __forceinline__ v16bf load_a_frag_f32(const float* __restrict__ ra,
                                                 const float* __restrict__ rb,
                                                 int koff) {
  v16bf a;
#pragma unroll
  for (int c = 0; c < 2; ++c) {
    const float* p = ra + koff + c * 16;
    float4 u0 = *(const float4*)p;
    float4 u1 = *(const float4*)(p + 4);
    if (SUM) {
      const float* q = rb + koff + c * 16;
      float4 s0 = *(const float4*)q;
      float4 s1 = *(const float4*)(q + 4);
      u0.x += s0.x; u0.y += s0.y; u0.z += s0.z; u0.w += s0.w;
      u1.x += s1.x; u1.y += s1.y; u1.z += s1.z; u1.w += s1.w;
    }
    const int o = c * 8;
    a[o + 0] = (__bf16)u0.x; a[o + 1] = (__bf16)u0.y;
    a[o + 2] = (__bf16)u0.z; a[o + 3] = (__bf16)u0.w;
    a[o + 4] = (__bf16)u1.x; a[o + 5] = (__bf16)u1.y;
    a[o + 6] = (__bf16)u1.z; a[o + 7] = (__bf16)u1.w;
  }
  return a;
}

// ---- A-fragment from bf16 row (e.g. LDS-staged intermediate) ---------------
__device__ __forceinline__ v16bf load_a_frag_bf16(const __bf16* __restrict__ row,
                                                  int koff) {
  bf16x8 c0 = *(const bf16x8*)(row + koff);
  bf16x8 c1 = *(const bf16x8*)(row + koff + 16);
  v16bf a;
#pragma unroll
  for (int e = 0; e < 8; ++e) { a[e] = c0[e]; a[e + 8] = c1[e]; }
  return a;
}

// ---- B-fragment (16-bit B 32x16 layout): 8-elem chunks at k0+8h, k0+16+8h ---
// wrow = &W_bf16[n][0] (W row-major (N,K)); B[k][n] = W[n][k].
__device__ __forceinline__ v16bf load_b_frag(const __bf16* __restrict__ wrow,
                                             int k0, int h) {
  const __bf16* p = wrow + k0 + 8 * h;
  bf16x8 c0 = *(const bf16x8*)p;
  bf16x8 c1 = *(const bf16x8*)(p + 16);
  v16bf b;
#pragma unroll
  for (int e = 0; e < 8; ++e) { b[e] = c0[e]; b[e + 8] = c1[e]; }
  return b;
}

// ---------------------------------------------------------------------------
// Generic GEMM: C[r][n] = act( (A1+A2?)[r][:] . Wbf[n][:] + bias[n] )
// 256 threads = 8 waves; 32 rows per block; wave = (rowTile, colGroup of N/4)
// ---------------------------------------------------------------------------
template <int K, int N, bool ADD2, bool RELU>
__global__ __launch_bounds__(256) void gemm_bias_kernel(
    const float* __restrict__ A1, const float* __restrict__ A2,
    const __bf16* __restrict__ Wbf, const float* __restrict__ bias,
    float* __restrict__ C) {
  constexpr int CT = N / 64;   // 16-wide col tiles per wave
  constexpr int KT = K / 32;
  const int lane = threadIdx.x & 31;
  const int wv   = threadIdx.x >> 5;
  const int rt = wv >> 2, cg = wv & 3;
  const int h = lane >> 4, l16 = lane & 15;
  const int rowBase = blockIdx.x * 32 + rt * 16;

  const float* aRow  = A1 + (size_t)(rowBase + l16) * K;
  const float* aRow2 = ADD2 ? (A2 + (size_t)(rowBase + l16) * K) : nullptr;

  v8f acc[CT] = {};
#pragma unroll
  for (int kt = 0; kt < KT; ++kt) {
    const int k0 = kt * 32;
    v16bf a = load_a_frag_f32<ADD2>(aRow, aRow2, k0 + 8 * h);
#pragma unroll
    for (int t = 0; t < CT; ++t) {
      const int n = cg * (N / 4) + t * 16 + l16;
      v16bf b = load_b_frag(Wbf + (size_t)n * K, k0, h);
      acc[t] = wmma_bf16(a, b, acc[t]);
    }
  }
#pragma unroll
  for (int t = 0; t < CT; ++t) {
    const int n = cg * (N / 4) + t * 16 + l16;
    const float bb = bias[n];
#pragma unroll
    for (int j = 0; j < 8; ++j) {
      float v = acc[t][j] + bb;
      if (RELU) v = fmaxf(v, 0.f);
      const int m = rowBase + 8 * h + j;
      C[(size_t)m * N + n] = v;
    }
  }
}

// ---------------------------------------------------------------------------
// Fused edge layer:  xe' = (x2 + (x1[ni,ii]+x1[no,jj]) @ mlpW^T + mlpB) @ edgeW^T + edgeB
// 32 rows/block of the (B*M*4, D) matrix; intermediate staged in LDS as bf16.
// Safe to run in-place on x2 (each block reads only its own rows first).
// ---------------------------------------------------------------------------
#define HSTR 264   // padded bf16 stride (256+8) to spread LDS banks
__global__ __launch_bounds__(256) void edge_layer_kernel(
    const float* __restrict__ x1,     // (B,T,2,D)
    const float* __restrict__ x2src,  // (B,M,4,D)
    float* __restrict__ x2dst,        // (B,M,4,D)
    const int* __restrict__ edgeIdx,  // 2*M  (ni row, no row)
    const __bf16* __restrict__ Wm, const float* __restrict__ bm,
    const __bf16* __restrict__ We, const float* __restrict__ be) {
  __shared__ __bf16 hTile[32][HSTR];

  const int lane = threadIdx.x & 31;
  const int wv   = threadIdx.x >> 5;
  const int rt = wv >> 2, cg = wv & 3;
  const int h = lane >> 4, l16 = lane & 15;
  const long blockRow = (long)blockIdx.x * 32;

  // My A-row (gathered): r = ((b*M + m)*4 + c)
  const long r   = blockRow + rt * 16 + l16;
  const int  c   = (int)(r & 3);
  const long bm_ = r >> 2;
  const int  b   = (int)(bm_ / MM);
  const int  m   = (int)(bm_ % MM);
  const int  ni  = edgeIdx[m];
  const int  no  = edgeIdx[MM + m];
  const float* ei = x1 + (((size_t)b * TT + ni) * 2 + (c >> 1)) * DD;
  const float* ej = x1 + (((size_t)b * TT + no) * 2 + (c & 1)) * DD;

  // ---- GEMM1: (ei+ej) @ mlpW^T ----
  v8f acc[4] = {};
#pragma unroll
  for (int kt = 0; kt < 8; ++kt) {
    const int k0 = kt * 32;
    v16bf a = load_a_frag_f32<true>(ei, ej, k0 + 8 * h);
#pragma unroll
    for (int t = 0; t < 4; ++t) {
      const int n = cg * 64 + t * 16 + l16;
      v16bf bfrg = load_b_frag(Wm + (size_t)n * DD, k0, h);
      acc[t] = wmma_bf16(a, bfrg, acc[t]);
    }
  }
  // epilogue: + mlp bias + x2 residual; stage as bf16 in LDS
#pragma unroll
  for (int t = 0; t < 4; ++t) {
    const int n = cg * 64 + t * 16 + l16;
    const float bb = bm[n];
#pragma unroll
    for (int j = 0; j < 8; ++j) {
      const int mrow = rt * 16 + 8 * h + j;
      const float v = acc[t][j] + bb + x2src[(size_t)(blockRow + mrow) * DD + n];
      hTile[mrow][n] = (__bf16)v;
    }
  }
  __syncthreads();

  // ---- GEMM2: h @ edgeW^T ----
  v8f acc2[4] = {};
  const __bf16* hrow = &hTile[rt * 16 + l16][0];
#pragma unroll
  for (int kt = 0; kt < 8; ++kt) {
    const int k0 = kt * 32;
    v16bf a = load_a_frag_bf16(hrow, k0 + 8 * h);
#pragma unroll
    for (int t = 0; t < 4; ++t) {
      const int n = cg * 64 + t * 16 + l16;
      v16bf bfrg = load_b_frag(We + (size_t)n * DD, k0, h);
      acc2[t] = wmma_bf16(a, bfrg, acc2[t]);
    }
  }
#pragma unroll
  for (int t = 0; t < 4; ++t) {
    const int n = cg * 64 + t * 16 + l16;
    const float bb = be[n];
#pragma unroll
    for (int j = 0; j < 8; ++j) {
      const int mrow = rt * 16 + 8 * h + j;
      x2dst[(size_t)(blockRow + mrow) * DD + n] = acc2[t][j] + bb;
    }
  }
}

// ---------------------------------------------------------------------------
// Scatter-add of edge aggregates into node accumulator (L2-resident, 8MB)
// ---------------------------------------------------------------------------
__global__ __launch_bounds__(256) void zero_kernel(float* __restrict__ p, long n) {
  const long i = (long)blockIdx.x * 256 + threadIdx.x;
  if (i < n) p[i] = 0.f;
}

__global__ __launch_bounds__(256) void scatter_kernel(
    const float* __restrict__ x2, const int* __restrict__ edgeIdx,
    float* __restrict__ acc) {
  const long i = (long)blockIdx.x * 256 + threadIdx.x;  // over B*M*D
  if (i >= (long)BB * MM * DD) return;
  const int  d  = (int)(i % DD);
  const long bm = i / DD;
  const int  m  = (int)(bm % MM);
  const int  b  = (int)(bm / MM);
  const float* e = x2 + bm * 4 * DD;
  const float v0 = 0.5f * (e[0 * DD + d] + e[1 * DD + d]);
  const float v1 = 0.5f * (e[2 * DD + d] + e[3 * DD + d]);
  const int ni = edgeIdx[m], no = edgeIdx[MM + m];
  float* ab = acc + (size_t)b * TT * 2 * DD;
  atomicAdd(ab + ((size_t)no * 2 + 0) * DD + d, v0);
  atomicAdd(ab + ((size_t)ni * 2 + 0) * DD + d, v0);
  atomicAdd(ab + ((size_t)no * 2 + 1) * DD + d, v1);
  atomicAdd(ab + ((size_t)ni * 2 + 1) * DD + d, v1);
}

// ---------------------------------------------------------------------------
// fp32 -> bf16 weight conversion (done once, ~2MB total)
// ---------------------------------------------------------------------------
__global__ __launch_bounds__(256) void cvt_bf16_kernel(
    const float* __restrict__ src, __bf16* __restrict__ dst, int n) {
  const int i = blockIdx.x * 256 + threadIdx.x;
  if (i < n) dst[i] = (__bf16)src[i];
}

// ---------------------------------------------------------------------------
// Final projection (N=1): per-wave dot of 256 + wave32 reduction
// ---------------------------------------------------------------------------
__global__ __launch_bounds__(256) void dot_kernel(
    const float* __restrict__ A, const float* __restrict__ w,
    const float* __restrict__ b, float* __restrict__ out, int rows) {
  const int wv = threadIdx.x >> 5, lane = threadIdx.x & 31;
  const int row = blockIdx.x * 8 + wv;
  if (row >= rows) return;
  const float* a = A + (size_t)row * 256;
  float s = 0.f;
#pragma unroll
  for (int i = 0; i < 8; ++i) s += a[lane + 32 * i] * w[lane + 32 * i];
#pragma unroll
  for (int off = 16; off > 0; off >>= 1) s += __shfl_xor(s, off, 32);
  if (lane == 0) out[row] = s + b[0];
}

// ---------------------------------------------------------------------------
extern "C" void kernel_launch(void* const* d_in, const int* in_sizes, int n_in,
                              void* d_out, int out_size, void* d_ws, size_t ws_size,
                              hipStream_t stream) {
  (void)in_sizes; (void)n_in; (void)out_size; (void)ws_size;

  const float* x1in  = (const float*)d_in[0];  // (B,T,2,D)
  const float* x2in  = (const float*)d_in[1];  // (B,M,4,D)
  const int*   edgeI = (const int*)d_in[2];    // (2,M)
  auto P = [&](int i) { return (const float*)d_in[i]; };
  // params: 3 + l*6 + {node_w,node_b,mlp_w,mlp_b,edge_w,edge_b}; 21..26 = r0..r2

  // ---- workspace carve ----
  char* ws = (char*)d_ws; size_t off = 0;
  auto take = [&](size_t bytes) -> void* {
    void* p = ws + off; off = (off + bytes + 255) & ~(size_t)255; return p;
  };
  __bf16* wbL[9];
  for (int i = 0; i < 9; ++i) wbL[i] = (__bf16*)take((size_t)256 * 256 * 2);
  __bf16* wbR0 = (__bf16*)take((size_t)512 * 512 * 2);
  __bf16* wbR1 = (__bf16*)take((size_t)256 * 512 * 2);
  float* acc = (float*)take((size_t)8192 * 256 * 4);   // (B,T,2,D)
  float* xnA = (float*)take((size_t)8192 * 256 * 4);
  float* xnB = (float*)take((size_t)8192 * 256 * 4);
  float* xe  = (float*)take((size_t)131072 * 256 * 4); // (B,M,4,D)
  float* h0  = (float*)take((size_t)4096 * 512 * 4);
  float* h1  = (float*)take((size_t)4096 * 256 * 4);

  // ---- weights -> bf16 (once) ----
  for (int l = 0; l < 3; ++l)
    for (int j = 0; j < 3; ++j)
      cvt_bf16_kernel<<<256, 256, 0, stream>>>(P(3 + l * 6 + j * 2), wbL[l * 3 + j], 65536);
  cvt_bf16_kernel<<<1024, 256, 0, stream>>>(P(21), wbR0, 262144);
  cvt_bf16_kernel<<<512, 256, 0, stream>>>(P(23), wbR1, 131072);

  // ---- GNN layers ----
  const float* xnSrc = x1in;
  float* xnDst = xnA;
  const float* xeSrc = x2in;
  const long accN = (long)8192 * 256;
  for (int l = 0; l < 3; ++l) {
    zero_kernel<<<(int)((accN + 255) / 256), 256, 0, stream>>>(acc, accN);
    scatter_kernel<<<(BB * MM * DD) / 256, 256, 0, stream>>>(xeSrc, edgeI, acc);
    if (l < 2)
      gemm_bias_kernel<256, 256, true, true><<<8192 / 32, 256, 0, stream>>>(
          xnSrc, acc, wbL[l * 3 + 0], P(3 + l * 6 + 1), xnDst);
    else
      gemm_bias_kernel<256, 256, true, false><<<8192 / 32, 256, 0, stream>>>(
          xnSrc, acc, wbL[l * 3 + 0], P(3 + l * 6 + 1), xnDst);
    if (l < 2) {  // layer-2 edge output is dead code in the reference
      edge_layer_kernel<<<131072 / 32, 256, 0, stream>>>(
          xnSrc, xeSrc, xe, edgeI,
          wbL[l * 3 + 1], P(3 + l * 6 + 3),
          wbL[l * 3 + 2], P(3 + l * 6 + 5));
      xeSrc = xe;  // subsequent layers run in-place on ws buffer
    }
    xnSrc = xnDst;
    xnDst = (xnDst == xnA) ? xnB : xnA;
  }

  // ---- readout: (4096,512) -> relu -> (4096,256) -> relu -> (4096,1) ----
  gemm_bias_kernel<512, 512, false, true><<<4096 / 32, 256, 0, stream>>>(
      xnSrc, nullptr, wbR0, P(22), h0);
  gemm_bias_kernel<512, 256, false, true><<<4096 / 32, 256, 0, stream>>>(
      h0, nullptr, wbR1, P(24), h1);
  dot_kernel<<<4096 / 8, 256, 0, stream>>>(h1, P(25), P(26), (float*)d_out, 4096);
}
